// Upsample_35682588295269
// MI455X (gfx1250) — matrix operand order repaired and checked by reading, not compiled, not run on hardware
//
#include <hip/hip_runtime.h>

typedef __attribute__((ext_vector_type(16))) _Float16 v16h;
typedef __attribute__((ext_vector_type(8)))  _Float16 v8h;
typedef __attribute__((ext_vector_type(8)))  float    v8f;

// ---------------------------------------------------------------------------
// Cast f32 weights -> f16 (layout [O, C*9] contiguous in w)
// ---------------------------------------------------------------------------
__global__ void cast_f32_to_f16(const float* __restrict__ in,
                                _Float16* __restrict__ out, int n) {
  int g = blockIdx.x * blockDim.x + threadIdx.x;
  if (g < n) out[g] = (_Float16)in[g];
}

// Cast + pad offset weights [27, Ktot] f32 -> [32, Ktot] f16 (rows 27..31 = 0)
__global__ void cast_pad_offw(const float* __restrict__ in,
                              _Float16* __restrict__ out, int Ktot) {
  int g = blockIdx.x * blockDim.x + threadIdx.x;
  int total = 32 * Ktot;
  if (g >= total) return;
  int row = g / Ktot, col = g - row * Ktot;
  out[g] = (row < 27) ? (_Float16)in[row * Ktot + col] : (_Float16)0.0f;
}

// ---------------------------------------------------------------------------
// Offset conv as im2col-GEMM on WMMA:
//   om[b, ch, p] = bias[ch] + sum_{c,k} w_off[ch, c*9+k] * im2col(x)[c*9+k, p]
// Workgroup: 256 thr = 8 waves; tile = 64 positions x 32 rows (27 valid).
// wave -> (M-tile = wv&1, N-subtile = wv>>1).
// ---------------------------------------------------------------------------
__global__ __launch_bounds__(256)
void offset_wmma_kernel(const float* __restrict__ x,
                        const _Float16* __restrict__ woq,  // [32, C*9] f16
                        const float* __restrict__ bias,
                        float* __restrict__ om,
                        int C, int H, int W) {
  const int HW = H * W;
  const int Ktot = C * 9;

  __shared__ int      s_voff[9 * 64];      // im2col offsets, -1 = zero pad
  __shared__ _Float16 s_b[2][64 * 32];     // double-buffered B tile [pos][K]

  int tid = threadIdx.x;
  int tilesPerBatch = HW / 64;
  int b  = blockIdx.x / tilesPerBatch;
  int p0 = (blockIdx.x % tilesPerBatch) * 64;

  // im2col gather table: shared across all C channels
  for (int t = tid; t < 9 * 64; t += 256) {
    int k = t / 64, pi = t - k * 64;
    int p = p0 + pi;
    int h = p / W, w = p - h * W;
    int yy = h + k / 3 - 1;
    int xx = w + k % 3 - 1;
    bool ok = (yy >= 0) && (yy < H) && (xx >= 0) && (xx < W);
    s_voff[t] = ok ? (yy * W + xx) : -1;
  }
  __syncthreads();

  int lane = tid & 31;
  int wv   = tid >> 5;
  int mt   = wv & 1;        // M tile: rows [16*mt, 16*mt+16)
  int nt   = wv >> 1;       // N subtile: positions [16*nt, 16*nt+16)
  int n    = lane & 15;
  int hh   = lane >> 4;

  v8f acc = {};
  const float* xb = x + (size_t)b * C * HW;
  int nChunks = Ktot / 32;

  for (int kc = 0; kc < nChunks; ++kc) {
    int buf = kc & 1;
    // build 32x64 f16 B tile: 2048 elems, 8/thread, coalesced over positions
#pragma unroll
    for (int e = 0; e < 8; ++e) {
      int el = e * 256 + tid;
      int pi = el & 63;
      int j  = el >> 6;               // 0..31
      int idx = kc * 32 + j;
      int c = idx / 9;
      int k = idx - c * 9;
      int off = s_voff[k * 64 + pi];
      float v = (off >= 0) ? xb[(size_t)c * HW + off] : 0.0f;
      s_b[buf][pi * 32 + j] = (_Float16)v;
    }
    __syncthreads();

    const _Float16* arow = woq + (size_t)(mt * 16 + n) * Ktot + kc * 32;
    v8h a0 = *(const v8h*)(arow + hh * 8);
    v8h a1 = *(const v8h*)(arow + 16 + hh * 8);
    const _Float16* brow = &s_b[buf][(nt * 16 + n) * 32 + hh * 8];
    v8h b0 = *(const v8h*)(brow);
    v8h b1 = *(const v8h*)(brow + 16);

    v16h av, bv;
#pragma unroll
    for (int i = 0; i < 8; ++i) {
      av[i] = a0[i]; av[8 + i] = a1[i];
      bv[i] = b0[i]; bv[8 + i] = b1[i];
    }
    acc = __builtin_amdgcn_wmma_f32_16x16x32_f16(
        false, av, false, bv, (short)0, acc, false, false);

    if (kc + 1 < nChunks) __builtin_prefetch(arow + 32, 0, 1);
  }

  // epilogue: bias add, store the 27 valid rows
#pragma unroll
  for (int r = 0; r < 8; ++r) {
    int ch = mt * 16 + hh * 8 + r;     // C/D layout: VGPR r -> M = r + 8*hh
    if (ch < 27) {
      om[((size_t)b * 27 + ch) * HW + p0 + nt * 16 + n] = acc[r] + bias[ch];
    }
  }
}

// ---------------------------------------------------------------------------
// Fused deformable-conv main stage:
//   bilinear gather (mask folded) -> GEMM [128 x C*9]x[C*9 x 16] via WMMA
//   -> bias + BN + ReLU.
// Workgroup: 256 thr = 8 waves; wave w owns output channels [16w, 16w+16),
// block owns 16 spatial positions of one batch image.
// ---------------------------------------------------------------------------
__global__ __launch_bounds__(256)
void dcn_wmma_kernel(const float* __restrict__ x, const float* __restrict__ om,
                     const _Float16* __restrict__ wq,
                     const float* __restrict__ bias,
                     const float* __restrict__ gamma,
                     const float* __restrict__ beta,
                     const float* __restrict__ mean,
                     const float* __restrict__ var,
                     float* __restrict__ out,
                     int C, int H, int W) {
  const int O = 128;
  const int HW = H * W;
  const int Ktot = C * 9;

  __shared__ float    s_wgt[9 * 16 * 4];   // bilinear weights (mask+valid folded)
  __shared__ int      s_off[9 * 16 * 4];   // clamped flat gather offsets
  __shared__ _Float16 s_b[2][16 * 32];     // double-buffered B tile [pos][K]

  int tid = threadIdx.x;
  int tilesPerBatch = HW / 16;
  int b  = blockIdx.x / tilesPerBatch;
  int p0 = (blockIdx.x % tilesPerBatch) * 16;

  // per-(k,pos) bilinear setup, shared across all C channels
  if (tid < 144) {
    int k = tid / 16, pi = tid % 16;
    int p = p0 + pi;
    int h = p / W, w = p % W;
    const float* omb = om + (size_t)b * 27 * HW + p;
    float oy = omb[(size_t)k * HW];
    float ox = omb[(size_t)(9 + k) * HW];
    float mz = omb[(size_t)(18 + k) * HW];
    float mask = 1.0f / (1.0f + __expf(-mz));

    float py = (float)h + (float)(k / 3 - 1) + oy;
    float px = (float)w + (float)(k % 3 - 1) + ox;
    float y0f = floorf(py), x0f = floorf(px);
    float fy = py - y0f, fx = px - x0f;
    int y0 = (int)y0f, x0 = (int)x0f;
    int y1 = y0 + 1, x1 = x0 + 1;
    float vy0 = (y0 >= 0 && y0 < H) ? 1.0f : 0.0f;
    float vy1 = (y1 >= 0 && y1 < H) ? 1.0f : 0.0f;
    float vx0 = (x0 >= 0 && x0 < W) ? 1.0f : 0.0f;
    float vx1 = (x1 >= 0 && x1 < W) ? 1.0f : 0.0f;
    int yc0 = min(max(y0, 0), H - 1), yc1 = min(max(y1, 0), H - 1);
    int xc0 = min(max(x0, 0), W - 1), xc1 = min(max(x1, 0), W - 1);

    float* sw = s_wgt + tid * 4;
    int*   so = s_off + tid * 4;
    sw[0] = (1.0f - fy) * (1.0f - fx) * vy0 * vx0 * mask;  so[0] = yc0 * W + xc0;
    sw[1] = (1.0f - fy) * fx          * vy0 * vx1 * mask;  so[1] = yc0 * W + xc1;
    sw[2] = fy * (1.0f - fx)          * vy1 * vx0 * mask;  so[2] = yc1 * W + xc0;
    sw[3] = fy * fx                   * vy1 * vx1 * mask;  so[3] = yc1 * W + xc1;
  }
  __syncthreads();

  int lane = tid & 31;
  int wv   = tid >> 5;      // wave id -> M tile
  int n    = lane & 15;     // column (position) / A row within tile
  int hh   = lane >> 4;     // lane half

  v8f acc = {};
  const float* xb = x + (size_t)b * C * HW;
  int nChunks = Ktot / 32;

  for (int kc = 0; kc < nChunks; ++kc) {
    int buf = kc & 1;
    // build 32x16 f16 B tile: gathered * bilinear weights, 2 elems/thread
#pragma unroll
    for (int e = 0; e < 2; ++e) {
      int el = tid + e * 256;
      int j  = el >> 4;                 // contract index within chunk, 0..31
      int pi = el & 15;                 // position within tile
      int idx = kc * 32 + j;            // global contract index = c*9+k
      int c = idx / 9;
      int k = idx - c * 9;
      const float* sw = s_wgt + (k * 16 + pi) * 4;
      const int*   so = s_off + (k * 16 + pi) * 4;
      const float* xp = xb + (size_t)c * HW;
      float v = sw[0] * xp[so[0]] + sw[1] * xp[so[1]] +
                sw[2] * xp[so[2]] + sw[3] * xp[so[3]];
      s_b[buf][pi * 32 + j] = (_Float16)v;
    }
    __syncthreads();

    // A fragment: 16x32 f16, lane half hh holds K in {g*16 + hh*8 + i}
    const _Float16* arow = wq + (size_t)(wv * 16 + n) * Ktot + kc * 32;
    v8h a0 = *(const v8h*)(arow + hh * 8);
    v8h a1 = *(const v8h*)(arow + 16 + hh * 8);
    // B fragment: 32x16 f16, lane half hh holds K in {g*16 + hh*8 + i}
    const _Float16* brow = &s_b[buf][n * 32 + hh * 8];
    v8h b0 = *(const v8h*)(brow);
    v8h b1 = *(const v8h*)(brow + 16);

    v16h av, bv;
#pragma unroll
    for (int i = 0; i < 8; ++i) {
      av[i] = a0[i]; av[8 + i] = a1[i];
      bv[i] = b0[i]; bv[8 + i] = b1[i];
    }
    acc = __builtin_amdgcn_wmma_f32_16x16x32_f16(
        false, av, false, bv, (short)0, acc, false, false);

    if (kc + 1 < nChunks) __builtin_prefetch(arow + 32, 0, 1);
  }

  // epilogue: (conv + bias) * inv + (beta - mean*inv), ReLU
#pragma unroll
  for (int r = 0; r < 8; ++r) {
    int o = wv * 16 + hh * 8 + r;          // C/D layout: VGPR r -> M = r + 8*hh
    float inv = gamma[o] * rsqrtf(var[o] + 1e-5f);
    float sh  = (bias[o] - mean[o]) * inv + beta[o];
    float v = acc[r] * inv + sh;
    v = v > 0.0f ? v : 0.0f;
    out[((size_t)b * O + o) * HW + p0 + n] = v;
  }
}

// ---------------------------------------------------------------------------
// Depthwise transposed conv, 4x4 kernel, stride 2 (lhs_dilation=2, pad 2,
// kernel pre-flipped in reference -> index with [3-ky][3-kx]).
// ---------------------------------------------------------------------------
__global__ void upsample_kernel(const float* __restrict__ y,
                                const float* __restrict__ wup,
                                float* __restrict__ out,
                                int Cc, int H, int W, int total) {
  int g = blockIdx.x * blockDim.x + threadIdx.x;
  if (g >= total) return;
  int H2 = H * 2, W2 = W * 2;
  int ox = g % W2;
  int r = g / W2;
  int oy = r % H2; r /= H2;
  int c = r % Cc;
  int b = r / Cc;

  const float* yp = y + ((size_t)b * Cc + c) * H * W;
  const float* wk = wup + c * 16;
  float acc = 0.0f;
#pragma unroll
  for (int ky = 0; ky < 4; ++ky) {
    int iy2 = oy + ky - 2;
    if (iy2 < 0 || (iy2 & 1)) continue;
    int iy = iy2 >> 1;
    if (iy >= H) continue;
#pragma unroll
    for (int kx = 0; kx < 4; ++kx) {
      int ix2 = ox + kx - 2;
      if (ix2 < 0 || (ix2 & 1)) continue;
      int ix = ix2 >> 1;
      if (ix >= W) continue;
      acc += wk[(3 - ky) * 4 + (3 - kx)] * yp[iy * W + ix];
    }
  }
  out[g] = acc;
}

// ---------------------------------------------------------------------------
extern "C" void kernel_launch(void* const* d_in, const int* in_sizes, int n_in,
                              void* d_out, int out_size, void* d_ws, size_t ws_size,
                              hipStream_t stream) {
  const float* x      = (const float*)d_in[0];
  const float* w_off1 = (const float*)d_in[1];
  const float* b_off1 = (const float*)d_in[2];
  const float* w1     = (const float*)d_in[3];
  const float* b1     = (const float*)d_in[4];
  const float* g1     = (const float*)d_in[5];
  const float* be1    = (const float*)d_in[6];
  const float* m1     = (const float*)d_in[7];
  const float* v1     = (const float*)d_in[8];
  const float* w_up   = (const float*)d_in[9];
  const float* w_off2 = (const float*)d_in[10];
  const float* b_off2 = (const float*)d_in[11];
  const float* w2     = (const float*)d_in[12];
  const float* b2     = (const float*)d_in[13];
  const float* g2     = (const float*)d_in[14];
  const float* be2    = (const float*)d_in[15];
  const float* m2     = (const float*)d_in[16];
  const float* v2     = (const float*)d_in[17];
  float* out = (float*)d_out;

  const int B = 4, C1 = 256, CO = 128, H1 = 64, W1 = 64, H2 = 128, W2 = 128;
  const int HW1 = H1 * W1, HW2 = H2 * W2;
  const int K1 = C1 * 9, K2 = CO * 9;

  char* ws = (char*)d_ws;
  size_t off = 0;
  auto alloc = [&](size_t bytes) -> void* {
    void* p = ws + off;
    off += (bytes + 255) & ~(size_t)255;
    return p;
  };
  _Float16* wq1  = (_Float16*)alloc((size_t)CO * K1 * sizeof(_Float16));
  _Float16* wq2  = (_Float16*)alloc((size_t)CO * K2 * sizeof(_Float16));
  _Float16* wo1q = (_Float16*)alloc((size_t)32 * K1 * sizeof(_Float16));
  _Float16* wo2q = (_Float16*)alloc((size_t)32 * K2 * sizeof(_Float16));
  float* om1 = (float*)alloc((size_t)B * 27 * HW1 * sizeof(float));
  float* om2 = (float*)alloc((size_t)B * 27 * HW2 * sizeof(float));
  float* y1  = (float*)alloc((size_t)B * CO * HW1 * sizeof(float));
  float* yup = (float*)alloc((size_t)B * CO * HW2 * sizeof(float));

  // weight casts (main f16 GEMM weights + zero-padded offset weights)
  {
    int n1 = CO * K1, n2 = CO * K2;
    cast_f32_to_f16<<<(n1 + 255) / 256, 256, 0, stream>>>(w1, wq1, n1);
    cast_f32_to_f16<<<(n2 + 255) / 256, 256, 0, stream>>>(w2, wq2, n2);
    int p1 = 32 * K1, p2 = 32 * K2;
    cast_pad_offw<<<(p1 + 255) / 256, 256, 0, stream>>>(w_off1, wo1q, K1);
    cast_pad_offw<<<(p2 + 255) / 256, 256, 0, stream>>>(w_off2, wo2q, K2);
  }
  // stage 1: WMMA offset conv + fused DCN + BN/ReLU
  offset_wmma_kernel<<<B * HW1 / 64, 256, 0, stream>>>(
      x, wo1q, b_off1, om1, C1, H1, W1);
  dcn_wmma_kernel<<<B * HW1 / 16, 256, 0, stream>>>(
      x, om1, wq1, b1, g1, be1, m1, v1, y1, C1, H1, W1);
  // upsample 2x depthwise transposed conv
  {
    int tot = B * CO * HW2;
    upsample_kernel<<<(tot + 255) / 256, 256, 0, stream>>>(
        y1, w_up, yup, CO, H1, W1, tot);
  }
  // stage 2: WMMA offset conv + fused DCN + BN/ReLU -> final output
  offset_wmma_kernel<<<B * HW2 / 64, 256, 0, stream>>>(
      yup, wo2q, b_off2, om2, CO, H2, W2);
  dcn_wmma_kernel<<<B * HW2 / 16, 256, 0, stream>>>(
      yup, om2, wq2, b2, g2, be2, m2, v2, out, CO, H2, W2);
}
